// EquiformerV2_89644557402689
// MI455X (gfx1250) — matrix operand order repaired and checked by reading, not compiled, not run on hardware
//
#include <hip/hip_runtime.h>
#include <hip/hip_bf16.h>

// ---------------------------------------------------------------------------
// EquiformerV2 block for MI455X (gfx1250, wave32, WMMA).
// All dense GEMMs run on v_wmma_f32_16x16x32_f16 (f16 in, f32 accumulate).
// Key simplification: v = h[src] @ Wv  ==  gather of P = h @ Wv  (node GEMM).
// ---------------------------------------------------------------------------

#define NNODES 10000
#define NEDGES 50000
#define NC     16      // (LMAX+1)^2
#define CCH    128
#define NH     8
#define DH     16
#define FFN    512
#define NB     600
#define NBP    608     // padded to multiple of 32 for WMMA K loop
#define LATD   256
#define NLAYER 2
#define CUTOFF 5.0f
#define AVG_DEG 3.946251153945923f
#define EPSV   1e-6f

typedef __attribute__((ext_vector_type(16))) _Float16 v16h;
typedef __attribute__((ext_vector_type(8)))  float    v8f;
typedef __attribute__((ext_vector_type(4)))  unsigned u32x4;

union FragU { v16h v; u32x4 q[2]; unsigned u[8]; };

// flags
#define GF_SILU 1
#define GF_ACC  2

// ---------------------------------------------------------------------------
// Generic WMMA GEMM:  C[M x Ncols] = A[M x K] (f16, lda) @ B (f16, stored
// transposed as Bt[Ncols x Kpad], ldb = Kpad).  8 waves -> 128 output cols
// per block; gridDim.y covers Ncols/128.  Epilogue: +bias, SiLU, accumulate,
// optional f16 side output.
// ---------------------------------------------------------------------------
__global__ void __launch_bounds__(256)
k_gemm_wmma(const _Float16* __restrict__ A, int lda,
            const _Float16* __restrict__ Bt, int ldb,
            const float* __restrict__ bias,
            float* __restrict__ Cf,
            _Float16* __restrict__ Ch,
            int ldc, int K, int flags)
{
    const int lane = threadIdx.x & 31;
    const int wave = threadIdx.x >> 5;
    const int r    = lane & 15;
    const int hi   = lane >> 4;

    const long m0 = (long)blockIdx.x * 16;
    const int  n0 = (blockIdx.y * 8 + wave) * 16;

    const _Float16* __restrict__ Arow = A  + (m0 + r) * (long)lda;
    const _Float16* __restrict__ Brow = Bt + (long)(n0 + r) * (long)ldb;

    v8f acc = {0.f,0.f,0.f,0.f,0.f,0.f,0.f,0.f};

    for (int k0 = 0; k0 < K; k0 += 32) {
        FragU a, b;
        // A 16x32 f16 fragment (ISA layout): lane-half hi holds K = {hi*8..hi*8+7, 16+hi*8..}
        a.q[0] = *(const u32x4*)(Arow + k0 + hi * 8);
        a.q[1] = *(const u32x4*)(Arow + k0 + 16 + hi * 8);
        // B 32x16 fragment (col r of tile): same K layout as A: K = {hi*8..hi*8+7, 16+hi*8..}
        b.q[0] = *(const u32x4*)(Brow + k0 + hi * 8);
        b.q[1] = *(const u32x4*)(Brow + k0 + 16 + hi * 8);
        acc = __builtin_amdgcn_wmma_f32_16x16x32_f16(
            false, a.v, false, b.v, (short)0, acc, false, false);
    }

    const float bv = bias ? bias[n0 + r] : 0.0f;
    const int   col = n0 + r;
#pragma unroll
    for (int g = 0; g < 8; ++g) {
        long row = m0 + g + hi * 8;
        float v  = acc[g] + bv;
        if (flags & GF_SILU) v = v / (1.0f + __expf(-v));
        long idx = row * (long)ldc + col;
        if (Cf) { if (flags & GF_ACC) Cf[idx] += v; else Cf[idx] = v; }
        if (Ch) Ch[idx] = (_Float16)v;
    }
}

// ---------------------------------------------------------------------------
// Small helper kernels
// ---------------------------------------------------------------------------

// f32 (K x Nc row-major) -> f16 transposed (Nc x Kp), zero-pad K..Kp
__global__ void k_convt(const float* __restrict__ src, _Float16* __restrict__ dst,
                        int K, int Kp, int Nc)
{
    long i = (long)blockIdx.x * blockDim.x + threadIdx.x;
    long total = (long)Nc * Kp;
    if (i >= total) return;
    int n = (int)(i / Kp), k = (int)(i % Kp);
    dst[i] = (k < K) ? (_Float16)src[(long)k * Nc + n] : (_Float16)0.0f;
}

// Xin[n, 0:128] = embed[an[n]]; Xin[n, 128:384] = latent[n]
__global__ void k_build_xin(const int* __restrict__ an,
                            const float* __restrict__ embed,
                            const float* __restrict__ latent,
                            _Float16* __restrict__ xin)
{
    long i = (long)blockIdx.x * blockDim.x + threadIdx.x;
    if (i >= (long)NNODES * 384) return;
    int n = (int)(i / 384), c = (int)(i % 384);
    float v = (c < CCH) ? embed[(long)an[n] * CCH + c]
                        : latent[(long)n * LATD + (c - CCH)];
    xin[i] = (_Float16)v;
}

// Gaussian smearing, padded to 608 columns, f16
__global__ void k_rbf(const float* __restrict__ dist, _Float16* __restrict__ rbf)
{
    long i = (long)blockIdx.x * blockDim.x + threadIdx.x;
    if (i >= (long)NEDGES * NBP) return;
    int e = (int)(i / NBP), b = (int)(i % NBP);
    if (b >= NB) { rbf[i] = (_Float16)0.0f; return; }
    const float step  = CUTOFF / (float)(NB - 1);
    const float width = 2.0f * step;
    const float coeff = -0.5f / (width * width);
    float d  = dist[e] * CUTOFF;
    float df = d - (float)b * step;
    rbf[i] = (_Float16)__expf(coeff * df * df);
}

// per-node RMS norm over (NC,C); writes f16 and/or f32
__global__ void k_rms(const float* __restrict__ x, const float* __restrict__ g,
                      _Float16* __restrict__ o16, float* __restrict__ o32)
{
    int n = blockIdx.x;
    const float* xr = x + (long)n * (NC * CCH);
    __shared__ float red[256];
    float s = 0.f;
    for (int i = threadIdx.x; i < NC * CCH; i += 256) { float v = xr[i]; s += v * v; }
    red[threadIdx.x] = s;
    __syncthreads();
    for (int w = 128; w > 0; w >>= 1) {
        if (threadIdx.x < w) red[threadIdx.x] += red[threadIdx.x + w];
        __syncthreads();
    }
    float scale = rsqrtf(red[0] / (float)(NC * CCH) + EPSV);
    for (int i = threadIdx.x; i < NC * CCH; i += 256) {
        float v = xr[i] * scale * g[i & (CCH - 1)];
        long idx = (long)n * (NC * CCH) + i;
        if (o16) o16[idx] = (_Float16)v;
        if (o32) o32[idx] = v;
    }
}

// x[:,0,:] += deg / AVG_DEG  (segment_sum by tgt via atomics)
__global__ void k_scatter_deg(const float* __restrict__ deg, const int* __restrict__ tgt,
                              float* __restrict__ x)
{
    long i = (long)blockIdx.x * blockDim.x + threadIdx.x;
    if (i >= (long)NEDGES * CCH) return;
    int e = (int)(i >> 7), c = (int)(i & (CCH - 1));
    atomicAdd(&x[(long)tgt[e] * (NC * CCH) + c], deg[i] * (1.0f / AVG_DEG));
}

// e16 = silu(S[src] + T[tgt] + r)
__global__ void k_edge_e(const float* __restrict__ S, const float* __restrict__ T,
                         const float* __restrict__ rb,
                         const int* __restrict__ src, const int* __restrict__ tgt,
                         _Float16* __restrict__ e16)
{
    long i = (long)blockIdx.x * blockDim.x + threadIdx.x;
    if (i >= (long)NEDGES * CCH) return;
    int e = (int)(i >> 7), c = (int)(i & (CCH - 1));
    float v = S[(long)src[e] * CCH + c] + T[(long)tgt[e] * CCH + c] + rb[i];
    e16[i] = (_Float16)(v / (1.0f + __expf(-v)));
}

__global__ void k_neg_inf(float* __restrict__ p, long n)
{
    long i = (long)blockIdx.x * blockDim.x + threadIdx.x;
    if (i < n) p[i] = -__builtin_inff();
}

__device__ inline void atomicMaxFloat(float* addr, float value)
{
    float old = *addr;
    while (old < value) {
        int assumed = __float_as_int(old);
        int prev = atomicCAS((int*)addr, assumed, __float_as_int(value));
        if (prev == assumed) break;
        old = __int_as_float(prev);
    }
}

// logits = e @ Walpha (128x8), plus segment max into segmax[tgt]
__global__ void k_logits(const _Float16* __restrict__ e16, const float* __restrict__ Wa,
                         const int* __restrict__ tgt,
                         float* __restrict__ logits, float* __restrict__ segmax)
{
    long i = (long)blockIdx.x * blockDim.x + threadIdx.x;
    if (i >= (long)NEDGES * NH) return;
    int e = (int)(i >> 3), h = (int)(i & (NH - 1));
    const _Float16* er = e16 + (long)e * CCH;
    float s = 0.f;
    for (int c = 0; c < CCH; ++c) s += (float)er[c] * Wa[c * NH + h];
    logits[i] = s;
    atomicMaxFloat(&segmax[(long)tgt[e] * NH + h], s);
}

// z = exp(logit - segmax[tgt]); denom[tgt] += z
__global__ void k_expz(float* __restrict__ z, const float* __restrict__ segmax,
                       const int* __restrict__ tgt, float* __restrict__ denom)
{
    long i = (long)blockIdx.x * blockDim.x + threadIdx.x;
    if (i >= (long)NEDGES * NH) return;
    int e = (int)(i >> 3), h = (int)(i & (NH - 1));
    float v = __expf(z[i] - segmax[(long)tgt[e] * NH + h]);
    z[i] = v;
    atomicAdd(&denom[(long)tgt[e] * NH + h], v);
}

// alpha = z / (denom[tgt] + eps)   (in place)
__global__ void k_alpha(float* __restrict__ z, const float* __restrict__ denom,
                        const int* __restrict__ tgt)
{
    long i = (long)blockIdx.x * blockDim.x + threadIdx.x;
    if (i >= (long)NEDGES * NH) return;
    int e = (int)(i >> 3), h = (int)(i & (NH - 1));
    z[i] = z[i] / (denom[(long)tgt[e] * NH + h] + EPSV);
}

// agg[tgt] += P[src] * alpha (broadcast over NC and DH)
__global__ void k_aggregate(const float* __restrict__ P, const float* __restrict__ alpha,
                            const int* __restrict__ src, const int* __restrict__ tgt,
                            float* __restrict__ agg)
{
    long i = (long)blockIdx.x * blockDim.x + threadIdx.x;
    if (i >= (long)NEDGES * NC * CCH) return;
    int e = (int)(i / (NC * CCH));
    int c = (int)(i % (NC * CCH));
    int h = (c >> 4) & (NH - 1);
    float v = P[(long)src[e] * (NC * CCH) + c] * alpha[(long)e * NH + h];
    atomicAdd(&agg[(long)tgt[e] * (NC * CCH) + c], v);
}

// f32 -> f16 flat copy
__global__ void k_f16copy(const float* __restrict__ s, _Float16* __restrict__ d, long n)
{
    long i = (long)blockIdx.x * blockDim.x + threadIdx.x;
    if (i < n) d[i] = (_Float16)s[i];
}

// ---------------------------------------------------------------------------
// Host side
// ---------------------------------------------------------------------------
static inline long cdivl(long a, long b) { return (a + b - 1) / b; }

extern "C" void kernel_launch(void* const* d_in, const int* in_sizes, int n_in,
                              void* d_out, int out_size, void* d_ws, size_t ws_size,
                              hipStream_t stream)
{
    (void)in_sizes; (void)n_in; (void)out_size; (void)ws_size;

    const int*   an      = (const int*)  d_in[0];
    const int*   eidx    = (const int*)  d_in[1];
    const float* edist   = (const float*)d_in[2];
    const float* latent  = (const float*)d_in[3];
    const float* embed   = (const float*)d_in[4];
    const float* latW    = (const float*)d_in[5];
    const float* latb    = (const float*)d_in[6];
    const float* degW1   = (const float*)d_in[7];
    const float* degW2   = (const float*)d_in[8];
    const float* Ws      = (const float*)d_in[9];
    const float* Wt      = (const float*)d_in[10];
    const float* Wr1     = (const float*)d_in[11];
    const float* Wr2     = (const float*)d_in[12];
    const float* Walpha  = (const float*)d_in[13];
    const float* Wv      = (const float*)d_in[14];
    const float* Wo      = (const float*)d_in[15];
    const float* W1      = (const float*)d_in[16];
    const float* W2      = (const float*)d_in[17];
    const float* g1      = (const float*)d_in[18];
    const float* g2      = (const float*)d_in[19];
    const float* gf      = (const float*)d_in[20];

    const int* src = eidx;
    const int* tgt = eidx + NEDGES;

    // ---- workspace bump allocator (256B aligned) ----
    char* base = (char*)d_ws;
    size_t off = 0;
    auto alloc = [&](size_t bytes) -> void* {
        off = (off + 255) & ~(size_t)255;
        void* p = base + off;
        off += bytes;
        return p;
    };

    _Float16* xin16   = (_Float16*)alloc((size_t)NNODES * 384 * 2);
    _Float16* rbf16   = (_Float16*)alloc((size_t)NEDGES * NBP * 2);
    _Float16* latWt   = (_Float16*)alloc((size_t)CCH * 384 * 2);
    _Float16* degW1t  = (_Float16*)alloc((size_t)CCH * NBP * 2);
    _Float16* degW2t  = (_Float16*)alloc((size_t)CCH * CCH * 2);
    _Float16* Wst[NLAYER], *Wtt[NLAYER], *Wr1t[NLAYER], *Wr2t[NLAYER];
    _Float16* Wvt[NLAYER], *Wot[NLAYER], *W1t[NLAYER],  *W2t[NLAYER];
    for (int l = 0; l < NLAYER; ++l) {
        Wst[l]  = (_Float16*)alloc((size_t)CCH * CCH * 2);
        Wtt[l]  = (_Float16*)alloc((size_t)CCH * CCH * 2);
        Wr1t[l] = (_Float16*)alloc((size_t)CCH * NBP * 2);
        Wr2t[l] = (_Float16*)alloc((size_t)CCH * CCH * 2);
        Wvt[l]  = (_Float16*)alloc((size_t)CCH * CCH * 2);
        Wot[l]  = (_Float16*)alloc((size_t)CCH * CCH * 2);
        W1t[l]  = (_Float16*)alloc((size_t)FFN * CCH * 2);
        W2t[l]  = (_Float16*)alloc((size_t)CCH * FFN * 2);
    }
    float*    x      = (float*)   alloc((size_t)NNODES * NC * CCH * 4);
    _Float16* h16    = (_Float16*)alloc((size_t)NNODES * NC * CCH * 2);
    float*    Sbuf   = (float*)   alloc((size_t)NNODES * CCH * 4);
    float*    Tbuf   = (float*)   alloc((size_t)NNODES * CCH * 4);
    _Float16* tE16   = (_Float16*)alloc((size_t)NEDGES * CCH * 2);
    float*    rBuf   = (float*)   alloc((size_t)NEDGES * CCH * 4);
    _Float16* e16    = (_Float16*)alloc((size_t)NEDGES * CCH * 2);
    float*    zBuf   = (float*)   alloc((size_t)NEDGES * NH * 4);
    float*    segmax = (float*)   alloc((size_t)NNODES * NH * 4);
    float*    denom  = (float*)   alloc((size_t)NNODES * NH * 4);
    float*    Pbuf   = (float*)   alloc((size_t)NNODES * NC * CCH * 4);
    float*    agg    = (float*)   alloc((size_t)NNODES * NC * CCH * 4);
    _Float16* tFFN16 = (_Float16*)alloc((size_t)NNODES * NC * FFN * 2);

    const int TB = 256;
    auto convt = [&](const float* s, _Float16* d, int K, int Kp, int Nc) {
        long tot = (long)Nc * Kp;
        k_convt<<<cdivl(tot, TB), TB, 0, stream>>>(s, d, K, Kp, Nc);
    };

    // ---- stage 0: conversions, embeddings, RBF ----
    hipMemsetAsync(x, 0, (size_t)NNODES * NC * CCH * 4, stream);

    k_build_xin<<<cdivl((long)NNODES * 384, TB), TB, 0, stream>>>(an, embed, latent, xin16);
    k_rbf<<<cdivl((long)NEDGES * NBP, TB), TB, 0, stream>>>(edist, rbf16);

    convt(latW,  latWt,  384, 384, CCH);
    convt(degW1, degW1t, NB,  NBP, CCH);
    convt(degW2, degW2t, CCH, CCH, CCH);
    for (int l = 0; l < NLAYER; ++l) {
        convt(Ws  + (size_t)l * CCH * CCH, Wst[l],  CCH, CCH, CCH);
        convt(Wt  + (size_t)l * CCH * CCH, Wtt[l],  CCH, CCH, CCH);
        convt(Wr1 + (size_t)l * NB  * CCH, Wr1t[l], NB,  NBP, CCH);
        convt(Wr2 + (size_t)l * CCH * CCH, Wr2t[l], CCH, CCH, CCH);
        convt(Wv  + (size_t)l * CCH * CCH, Wvt[l],  CCH, CCH, CCH);
        convt(Wo  + (size_t)l * CCH * CCH, Wot[l],  CCH, CCH, CCH);
        convt(W1  + (size_t)l * CCH * FFN, W1t[l],  CCH, CCH, FFN);
        convt(W2  + (size_t)l * FFN * CCH, W2t[l],  FFN, FFN, CCH);
    }

    // x[:,0,:] = concat(embed, latent) @ latW + b   (ldc = NC*C picks nc=0 rows)
    k_gemm_wmma<<<dim3(NNODES / 16, 1), TB, 0, stream>>>(
        xin16, 384, latWt, 384, latb, x, nullptr, NC * CCH, 384, 0);

    // deg = silu(rbf @ degW1) @ degW2 ; x[:,0,:] += segsum(deg)/AVG_DEG
    k_gemm_wmma<<<dim3(NEDGES / 16, 1), TB, 0, stream>>>(
        rbf16, NBP, degW1t, NBP, nullptr, nullptr, tE16, CCH, NBP, GF_SILU);
    k_gemm_wmma<<<dim3(NEDGES / 16, 1), TB, 0, stream>>>(
        tE16, CCH, degW2t, CCH, nullptr, rBuf, nullptr, CCH, CCH, 0);
    k_scatter_deg<<<cdivl((long)NEDGES * CCH, TB), TB, 0, stream>>>(rBuf, tgt, x);

    // ---- transformer layers ----
    for (int l = 0; l < NLAYER; ++l) {
        // h = rms_norm(x, g1[l]) -> f16
        k_rms<<<NNODES, TB, 0, stream>>>(x, g1 + (size_t)l * CCH, h16, nullptr);

        // S = inv @ Ws ; T = inv @ Wt  (inv = h[:,0,:]: row stride NC*C)
        k_gemm_wmma<<<dim3(NNODES / 16, 1), TB, 0, stream>>>(
            h16, NC * CCH, Wst[l], CCH, nullptr, Sbuf, nullptr, CCH, CCH, 0);
        k_gemm_wmma<<<dim3(NNODES / 16, 1), TB, 0, stream>>>(
            h16, NC * CCH, Wtt[l], CCH, nullptr, Tbuf, nullptr, CCH, CCH, 0);

        // r = silu(rbf @ Wr1) @ Wr2
        k_gemm_wmma<<<dim3(NEDGES / 16, 1), TB, 0, stream>>>(
            rbf16, NBP, Wr1t[l], NBP, nullptr, nullptr, tE16, CCH, NBP, GF_SILU);
        k_gemm_wmma<<<dim3(NEDGES / 16, 1), TB, 0, stream>>>(
            tE16, CCH, Wr2t[l], CCH, nullptr, rBuf, nullptr, CCH, CCH, 0);

        // e = silu(S[src] + T[tgt] + r)
        k_edge_e<<<cdivl((long)NEDGES * CCH, TB), TB, 0, stream>>>(
            Sbuf, Tbuf, rBuf, src, tgt, e16);

        // segment softmax over tgt
        k_neg_inf<<<cdivl((long)NNODES * NH, TB), TB, 0, stream>>>(segmax, (long)NNODES * NH);
        hipMemsetAsync(denom, 0, (size_t)NNODES * NH * 4, stream);
        k_logits<<<cdivl((long)NEDGES * NH, TB), TB, 0, stream>>>(
            e16, Walpha + (size_t)l * CCH * NH, tgt, zBuf, segmax);
        k_expz<<<cdivl((long)NEDGES * NH, TB), TB, 0, stream>>>(zBuf, segmax, tgt, denom);
        k_alpha<<<cdivl((long)NEDGES * NH, TB), TB, 0, stream>>>(zBuf, denom, tgt);

        // P = h @ Wv (all NC rows);  agg[tgt] += P[src] * alpha
        k_gemm_wmma<<<dim3((NNODES * NC) / 16, 1), TB, 0, stream>>>(
            h16, CCH, Wvt[l], CCH, nullptr, Pbuf, nullptr, CCH, CCH, 0);
        hipMemsetAsync(agg, 0, (size_t)NNODES * NC * CCH * 4, stream);
        k_aggregate<<<cdivl((long)NEDGES * NC * CCH, TB), TB, 0, stream>>>(
            Pbuf, zBuf, src, tgt, agg);

        // x += agg @ Wo   (agg -> f16 reusing h16)
        k_f16copy<<<cdivl((long)NNODES * NC * CCH, TB), TB, 0, stream>>>(
            agg, h16, (long)NNODES * NC * CCH);
        k_gemm_wmma<<<dim3((NNODES * NC) / 16, 1), TB, 0, stream>>>(
            h16, CCH, Wot[l], CCH, nullptr, x, nullptr, CCH, CCH, GF_ACC);

        // FFN: x += silu(rms(x,g2) @ W1) @ W2
        k_rms<<<NNODES, TB, 0, stream>>>(x, g2 + (size_t)l * CCH, h16, nullptr);
        k_gemm_wmma<<<dim3((NNODES * NC) / 16, FFN / 128), TB, 0, stream>>>(
            h16, CCH, W1t[l], CCH, nullptr, nullptr, tFFN16, FFN, CCH, GF_SILU);
        k_gemm_wmma<<<dim3((NNODES * NC) / 16, 1), TB, 0, stream>>>(
            tFFN16, FFN, W2t[l], FFN, nullptr, x, nullptr, CCH, FFN, GF_ACC);
    }

    // final rms_norm(x, gf) -> d_out (fp32)
    k_rms<<<NNODES, TB, 0, stream>>>(x, gf, nullptr, (float*)d_out);
}
